// BiGRUModel_27315992002657
// MI455X (gfx1250) — compile-verified, matrix operand order checked by reading, not run on hardware
//
#include <hip/hip_runtime.h>

#define TSEQ 2048
#define BATCH 256
#define HID 64
#define GD 192     // 3*HID
#define D2 128     // 2*HID

typedef __bf16 bf16_t;
typedef bf16_t v16bf __attribute__((ext_vector_type(16)));
typedef float  v8f   __attribute__((ext_vector_type(8)));

union FragU {
    v16bf v;
    unsigned short s[16];
    uint4 q[2];
};

__device__ __forceinline__ unsigned short f2bf(float f) {
    unsigned int u = __float_as_uint(f);
    u += 0x7FFFu + ((u >> 16) & 1u);   // round-to-nearest-even
    return (unsigned short)(u >> 16);
}

__device__ __forceinline__ v8f wmma_bf16(v16bf a, v16bf b, v8f c) {
    return __builtin_amdgcn_wmma_f32_16x16x32_bf16(false, a, false, b, (short)0, c, false, false);
}

// Guaranteed-b128 LDS fragment load (32B aligned slots).
__device__ __forceinline__ v16bf ldfrag(const unsigned short* p) {
    FragU f;
    const uint4* q = (const uint4*)__builtin_assume_aligned(p, 32);
    f.q[0] = q[0];
    f.q[1] = q[1];
    return f.v;
}

// Branch-free fast activations: v_exp_f32 + v_rcp_f32, no IEEE div, no EXEC divergence.
__device__ __forceinline__ float sigm(float x) {
    return __builtin_amdgcn_rcpf(1.0f + __expf(-x));
}
__device__ __forceinline__ float tanh_fast(float x) {
    return 2.0f * __builtin_amdgcn_rcpf(1.0f + __expf(-2.0f * x)) - 1.0f;
}

// A-fragment slot map for 16x32 bf16 tiles:
//   element (row m, K k): c = k>>5; kk = k&31; lane = m + 16*((kk>>3)&1);
//   idx = (kk&7) + 8*((kk>>4)&1);  stored at frag[c][lane][idx]

// ---------------------------------------------------------------------------
// Layer 0: bidirectional GRU over x[B,T,8].  grid=32 (dir*16 + batch-tile),
// block=128 (4 waves; wave w owns hidden slice [16w,16w+16)).
// All weight fragments are register-resident; LDS only carries the h state.
// Writes h0out bf16 [B][T][128] (fwd cols 0..63, bwd cols 64..127).
// ---------------------------------------------------------------------------
__global__ __launch_bounds__(128) void k_gru_l0(
    const float* __restrict__ x,
    const float* __restrict__ WihF, const float* __restrict__ WhhF,
    const float* __restrict__ bihF, const float* __restrict__ bhhF,
    const float* __restrict__ WihB, const float* __restrict__ WhhB,
    const float* __restrict__ bihB, const float* __restrict__ bhhB,
    unsigned short* __restrict__ h0out)
{
    __shared__ __attribute__((aligned(32))) unsigned short apackedH[2 * 32 * 16]; // h as A-frags
    __shared__ __attribute__((aligned(16))) unsigned short hbuf[16 * 64];         // h linear for coalesced store

    const int tid  = threadIdx.x;
    const int w    = tid >> 5;
    const int l    = tid & 31;
    const int lm   = l & 15;
    const int hi   = l >> 4;
    const int koff = hi * 8;
    const int dir  = blockIdx.x >> 4;
    const int b0   = (blockIdx.x & 15) * 16;

    const float* Wih = dir ? WihB : WihF;
    const float* Whh = dir ? WhhB : WhhF;
    const float* bih = dir ? bihB : bihF;
    const float* bhh = dir ? bhhB : bhhF;

    for (int e = tid; e < 2 * 512; e += 128) apackedH[e] = 0;  // h(-1) = 0

    // Register-resident Whh^T fragments (3 gate tiles x 2 K-chunks per wave).
    FragU ghf[3][2];
    #pragma unroll
    for (int g = 0; g < 3; ++g) {
        int nt = g * 4 + w;
        int n  = nt * 16 + lm;
        #pragma unroll
        for (int c = 0; c < 2; ++c)
            #pragma unroll
            for (int idx = 0; idx < 16; ++idx) {
                int k = c * 32 + koff + (idx & 7) + 16 * (idx >> 3);
                ghf[g][c].s[idx] = f2bf(Whh[n * HID + k]);
            }
    }
    // Register-resident Wih^T fragments (K=8 real, padded to 32).
    FragU gif[3];
    #pragma unroll
    for (int g = 0; g < 3; ++g) {
        int nt = g * 4 + w;
        int n  = nt * 16 + lm;
        #pragma unroll
        for (int idx = 0; idx < 16; ++idx) {
            int k = koff + (idx & 7) + 16 * (idx >> 3);
            gif[g].s[idx] = (k < 8) ? f2bf(Wih[n * 8 + k]) : (unsigned short)0;
        }
    }

    // Per-lane biases for this wave's hidden slice.
    const int   nr   = 16 * w + lm;
    const float b_r  = bih[nr] + bhh[nr];
    const float b_z  = bih[64 + nr] + bhh[64 + nr];
    const float b_ni = bih[128 + nr];
    const float b_nh = bhh[128 + nr];

    // Precomputed h-write slot (k = nr, fixed per lane).
    const int jc   = nr >> 5;
    const int jkk  = nr & 31;
    const int jkhi = (jkk >> 3) & 1;
    const int jidx = (jkk & 7) + 8 * ((jkk >> 4) & 1);

    v8f hstate;
    #pragma unroll
    for (int i = 0; i < 8; ++i) hstate[i] = 0.f;

    __syncthreads();

    #pragma unroll 1
    for (int s = 0; s < TSEQ; ++s) {
        const int t = dir ? (TSEQ - 1 - s) : s;

        // A fragment of x_t (16x32, K<8 valid, rest zero).
        FragU ax;
        #pragma unroll
        for (int i = 0; i < 16; ++i) ax.s[i] = 0;
        if (l < 16) {
            const float* xp = x + ((size_t)(b0 + lm) * TSEQ + t) * 8;
            #pragma unroll
            for (int i = 0; i < 8; ++i) ax.s[i] = f2bf(xp[i]);
        }

        // A fragments of h (packed -> contiguous b128 loads).
        v16bf ah0 = ldfrag(&apackedH[(0 * 32 + l) * 16]);
        v16bf ah1 = ldfrag(&apackedH[(1 * 32 + l) * 16]);

        __syncthreads();   // all reads done before this step's LDS writes

        v8f ar, az, ani, anh;
        #pragma unroll
        for (int i = 0; i < 8; ++i) { ar[i] = b_r; az[i] = b_z; ani[i] = b_ni; anh[i] = b_nh; }

        // gi = x @ Wih^T   (r->tile w, z->4+w, n->8+w); all operands in VGPRs
        ar  = wmma_bf16(ax.v, gif[0].v, ar);
        az  = wmma_bf16(ax.v, gif[1].v, az);
        ani = wmma_bf16(ax.v, gif[2].v, ani);

        // gh = h @ Whh^T
        ar  = wmma_bf16(ah0, ghf[0][0].v, ar);
        ar  = wmma_bf16(ah1, ghf[0][1].v, ar);
        az  = wmma_bf16(ah0, ghf[1][0].v, az);
        az  = wmma_bf16(ah1, ghf[1][1].v, az);
        anh = wmma_bf16(ah0, ghf[2][0].v, anh);
        anh = wmma_bf16(ah1, ghf[2][1].v, anh);

        // GRU gates + state update; stage next-step A frags and output row.
        #pragma unroll
        for (int i = 0; i < 8; ++i) {
            float r  = sigm(ar[i]);
            float z  = sigm(az[i]);
            float n  = tanh_fast(ani[i] + r * anh[i]);
            float hn = (1.f - z) * n + z * hstate[i];
            hstate[i] = hn;
            unsigned short hb = f2bf(hn);
            int m = i + 8 * hi;
            hbuf[m * 64 + nr] = hb;
            apackedH[(jc * 32 + (m + 16 * jkhi)) * 16 + jidx] = hb;
        }

        __syncthreads();   // LDS writes visible

        // Coalesced b128 store of h_t into h0out[:, t, dir*64 + :].
        {
            int   fd = tid * 4;           // dword index into hbuf
            int   m  = fd >> 5;
            int   dd = fd & 31;
            uint4 vv = *(const uint4*)__builtin_assume_aligned(&hbuf[fd * 2], 16);
            *(uint4*)&h0out[((size_t)(b0 + m) * TSEQ + t) * D2 + dir * HID + dd * 2] = vv;
        }
    }
}

// ---------------------------------------------------------------------------
// Layer 1 forward recurrence over h0[B,T,128]; only final state kept.
// grid=16 (batch tiles), block=128.  Writes h1f f32 [B][64].
// ---------------------------------------------------------------------------
__global__ __launch_bounds__(128) void k_gru_l1f(
    const unsigned short* __restrict__ h0,
    const float* __restrict__ Wih, const float* __restrict__ Whh,
    const float* __restrict__ bih, const float* __restrict__ bhh,
    float* __restrict__ h1f)
{
    __shared__ __attribute__((aligned(32))) unsigned short fragGi[12 * 4 * 32 * 16]; // 48KB Wih^T frags
    __shared__ __attribute__((aligned(32))) unsigned short apackedX[4 * 32 * 16];    // x_t as A-frags
    __shared__ __attribute__((aligned(32))) unsigned short apackedH[2 * 32 * 16];    // h as A-frags

    const int tid  = threadIdx.x;
    const int w    = tid >> 5;
    const int l    = tid & 31;
    const int lm   = l & 15;
    const int hi   = l >> 4;
    const int koff = hi * 8;
    const int b0   = blockIdx.x * 16;

    // Pack Wih^T fragments: frag (nt 0..11, c 0..3).
    for (int e = tid; e < 12 * 4 * 512; e += 128) {
        int frag = e >> 9, lane = (e >> 4) & 31, idx = e & 15;
        int nt = frag >> 2, c = frag & 3;
        int n  = nt * 16 + (lane & 15);
        int ko = (lane < 16) ? 0 : 8;
        int k  = c * 32 + ko + (idx & 7) + 16 * (idx >> 3);
        fragGi[e] = f2bf(Wih[n * D2 + k]);
    }
    for (int e = tid; e < 2 * 512; e += 128) apackedH[e] = 0;

    // This wave's Whh^T fragments live in registers (3 gate tiles x 2 K-chunks).
    FragU ghf[3][2];
    #pragma unroll
    for (int g = 0; g < 3; ++g) {
        int nt = g * 4 + w;
        int n  = nt * 16 + lm;
        #pragma unroll
        for (int c = 0; c < 2; ++c)
            #pragma unroll
            for (int idx = 0; idx < 16; ++idx) {
                int k = c * 32 + koff + (idx & 7) + 16 * (idx >> 3);
                ghf[g][c].s[idx] = f2bf(Whh[n * HID + k]);
            }
    }

    const int   nr   = 16 * w + lm;
    const float b_r  = bih[nr] + bhh[nr];
    const float b_z  = bih[64 + nr] + bhh[64 + nr];
    const float b_ni = bih[128 + nr];
    const float b_nh = bhh[128 + nr];

    const int jc   = nr >> 5;
    const int jkk  = nr & 31;
    const int jkhi = (jkk >> 3) & 1;
    const int jidx = (jkk & 7) + 8 * ((jkk >> 4) & 1);

    v8f hstate;
    #pragma unroll
    for (int i = 0; i < 8; ++i) hstate[i] = 0.f;

    __syncthreads();

    #pragma unroll 1
    for (int t = 0; t < TSEQ; ++t) {
        // Stage h0[:, t, :] (16x128 bf16) directly into A-fragment order.
        #pragma unroll
        for (int i = 0; i < 4; ++i) {
            int fd = (i * 128 + tid) * 2;       // even dword index 0..1022
            int m  = fd >> 6;
            int dd = fd & 63;
            uint2 vv = *(const uint2*)(h0 + ((size_t)(b0 + m) * TSEQ + t) * D2 + dd * 2);
            int c    = dd >> 4;
            int slot = (dd & 3) + 4 * ((dd >> 3) & 1);
            int khi  = (dd >> 2) & 1;
            *(uint2*)&apackedX[((c * 32 + (m + 16 * khi)) * 16) + slot * 2] = vv;
        }
        if (t + 4 < TSEQ)   // pull a future timestep toward L2/WGP$
            __builtin_prefetch(h0 + ((size_t)(b0 + (tid & 15)) * TSEQ + (t + 4)) * D2, 0, 1);

        __syncthreads();    // staged tile visible

        v16bf axv[4];
        #pragma unroll
        for (int c = 0; c < 4; ++c) axv[c] = ldfrag(&apackedX[(c * 32 + l) * 16]);
        v16bf ah0 = ldfrag(&apackedH[(0 * 32 + l) * 16]);
        v16bf ah1 = ldfrag(&apackedH[(1 * 32 + l) * 16]);

        __syncthreads();    // reads done before this step's writes

        v8f ar, az, ani, anh;
        #pragma unroll
        for (int i = 0; i < 8; ++i) { ar[i] = b_r; az[i] = b_z; ani[i] = b_ni; anh[i] = b_nh; }

        // gi = h0_t @ Wih^T  (4 K-chunks per gate tile)
        #pragma unroll
        for (int c = 0; c < 4; ++c) {
            ar  = wmma_bf16(axv[c], ldfrag(&fragGi[(((w    ) * 4 + c) * 32 + l) * 16]), ar);
            az  = wmma_bf16(axv[c], ldfrag(&fragGi[(((4 + w) * 4 + c) * 32 + l) * 16]), az);
            ani = wmma_bf16(axv[c], ldfrag(&fragGi[(((8 + w) * 4 + c) * 32 + l) * 16]), ani);
        }
        // gh = h @ Whh^T  (register-resident fragments)
        ar  = wmma_bf16(ah0, ghf[0][0].v, ar);
        ar  = wmma_bf16(ah1, ghf[0][1].v, ar);
        az  = wmma_bf16(ah0, ghf[1][0].v, az);
        az  = wmma_bf16(ah1, ghf[1][1].v, az);
        anh = wmma_bf16(ah0, ghf[2][0].v, anh);
        anh = wmma_bf16(ah1, ghf[2][1].v, anh);

        #pragma unroll
        for (int i = 0; i < 8; ++i) {
            float r  = sigm(ar[i]);
            float z  = sigm(az[i]);
            float n  = tanh_fast(ani[i] + r * anh[i]);
            float hn = (1.f - z) * n + z * hstate[i];
            hstate[i] = hn;
            int m = i + 8 * hi;
            apackedH[(jc * 32 + (m + 16 * jkhi)) * 16 + jidx] = f2bf(hn);
        }
    }

    #pragma unroll
    for (int i = 0; i < 8; ++i)
        h1f[(size_t)(b0 + i + 8 * hi) * HID + nr] = hstate[i];
}

// ---------------------------------------------------------------------------
// Head: layer-1 backward is a single GRU step from zero state at t=T-1,
// then last=[h1f,h1b], hid=relu(last@W1^T+b1), logits=hid@W2^T+b2.
// grid=16, block=128.
// ---------------------------------------------------------------------------
__global__ __launch_bounds__(128) void k_head(
    const unsigned short* __restrict__ h0,
    const float* __restrict__ h1f,
    const float* __restrict__ WihB, const float* __restrict__ bihB, const float* __restrict__ bhhB,
    const float* __restrict__ W1, const float* __restrict__ b1,
    const float* __restrict__ W2, const float* __restrict__ b2,
    float* __restrict__ out)
{
    __shared__ __attribute__((aligned(32))) unsigned short fragGi[12 * 4 * 32 * 16]; // Wih_l1_b^T frags
    __shared__ __attribute__((aligned(32))) unsigned short apackedX[4 * 32 * 16];    // h0[:,T-1,:]
    __shared__ __attribute__((aligned(32))) unsigned short apackedL[4 * 32 * 16];    // last (16x128)
    __shared__ __attribute__((aligned(32))) unsigned short apackedM[4 * 32 * 16];    // hidden (16x128)

    const int tid  = threadIdx.x;
    const int w    = tid >> 5;
    const int l    = tid & 31;
    const int lm   = l & 15;
    const int hi   = l >> 4;
    const int koff = hi * 8;
    const int b0   = blockIdx.x * 16;

    // Pack Wih_l1_b^T fragments.
    for (int e = tid; e < 12 * 4 * 512; e += 128) {
        int frag = e >> 9, lane = (e >> 4) & 31, idx = e & 15;
        int nt = frag >> 2, c = frag & 3;
        int n  = nt * 16 + (lane & 15);
        int ko = (lane < 16) ? 0 : 8;
        int k  = c * 32 + ko + (idx & 7) + 16 * (idx >> 3);
        fragGi[e] = f2bf(WihB[n * D2 + k]);
    }

    // W1^T fragments in registers: this wave's 2 output tiles {w, w+4}.
    FragU w1f[2][4];
    #pragma unroll
    for (int ti = 0; ti < 2; ++ti) {
        int nt = w + ti * 4;
        int n  = nt * 16 + lm;
        #pragma unroll
        for (int c = 0; c < 4; ++c)
            #pragma unroll
            for (int idx = 0; idx < 16; ++idx) {
                int k = c * 32 + koff + (idx & 7) + 16 * (idx >> 3);
                w1f[ti][c].s[idx] = f2bf(W1[n * D2 + k]);
            }
    }
    // W2^T fragments (only waves 0/1 use them; N padded 24->32).
    FragU w2f[4];
    {
        int n = (w & 1) * 16 + lm;
        #pragma unroll
        for (int c = 0; c < 4; ++c)
            #pragma unroll
            for (int idx = 0; idx < 16; ++idx) {
                int k = c * 32 + koff + (idx & 7) + 16 * (idx >> 3);
                w2f[c].s[idx] = (n < 24) ? f2bf(W2[n * D2 + k]) : (unsigned short)0;
            }
    }

    // Stage h0[:, T-1, :] into A-fragment order.
    #pragma unroll
    for (int i = 0; i < 4; ++i) {
        int fd = (i * 128 + tid) * 2;
        int m  = fd >> 6;
        int dd = fd & 63;
        uint2 vv = *(const uint2*)(h0 + ((size_t)(b0 + m) * TSEQ + (TSEQ - 1)) * D2 + dd * 2);
        int c    = dd >> 4;
        int slot = (dd & 3) + 4 * ((dd >> 3) & 1);
        int khi  = (dd >> 2) & 1;
        *(uint2*)&apackedX[((c * 32 + (m + 16 * khi)) * 16) + slot * 2] = vv;
    }
    // Stage h1f (forward final state) into last[:, 0:64].
    for (int e = tid; e < 1024; e += 128) {
        int m = e >> 6, j = e & 63;
        unsigned short vb = f2bf(h1f[(size_t)(b0 + m) * HID + j]);
        int kk = j & 31;
        apackedL[(((j >> 5) * 32 + (m + 16 * ((kk >> 3) & 1))) * 16) + (kk & 7) + 8 * ((kk >> 4) & 1)] = vb;
    }
    __syncthreads();

    // Layer-1 backward: one GRU step from h=0 (gh == bhh).
    {
        v16bf axv[4];
        #pragma unroll
        for (int c = 0; c < 4; ++c) axv[c] = ldfrag(&apackedX[(c * 32 + l) * 16]);

        const int   nr   = 16 * w + lm;
        const float b_r  = bihB[nr] + bhhB[nr];
        const float b_z  = bihB[64 + nr] + bhhB[64 + nr];
        const float b_ni = bihB[128 + nr];
        const float b_nh = bhhB[128 + nr];

        v8f ar, az, ani;
        #pragma unroll
        for (int i = 0; i < 8; ++i) { ar[i] = b_r; az[i] = b_z; ani[i] = b_ni; }

        #pragma unroll
        for (int c = 0; c < 4; ++c) {
            ar  = wmma_bf16(axv[c], ldfrag(&fragGi[(((w    ) * 4 + c) * 32 + l) * 16]), ar);
            az  = wmma_bf16(axv[c], ldfrag(&fragGi[(((4 + w) * 4 + c) * 32 + l) * 16]), az);
            ani = wmma_bf16(axv[c], ldfrag(&fragGi[(((8 + w) * 4 + c) * 32 + l) * 16]), ani);
        }

        const int j2   = 64 + nr;
        const int jc   = j2 >> 5;
        const int jkk  = j2 & 31;
        const int jkhi = (jkk >> 3) & 1;
        const int jidx = (jkk & 7) + 8 * ((jkk >> 4) & 1);
        #pragma unroll
        for (int i = 0; i < 8; ++i) {
            float r  = sigm(ar[i]);
            float z  = sigm(az[i]);
            float n  = tanh_fast(ani[i] + r * b_nh);
            float hb = (1.f - z) * n;           // h_prev == 0
            int m = i + 8 * hi;
            apackedL[(jc * 32 + (m + 16 * jkhi)) * 16 + jidx] = f2bf(hb);
        }
    }
    __syncthreads();

    // hid = relu(last @ W1^T + b1): wave w computes tiles {w, w+4}.
    {
        v16bf alv[4];
        #pragma unroll
        for (int c = 0; c < 4; ++c) alv[c] = ldfrag(&apackedL[(c * 32 + l) * 16]);

        #pragma unroll
        for (int ti = 0; ti < 2; ++ti) {
            int   nt = w + ti * 4;
            int   nn = nt * 16 + lm;
            float bb = b1[nn];
            v8f acc;
            #pragma unroll
            for (int i = 0; i < 8; ++i) acc[i] = bb;
            #pragma unroll
            for (int c = 0; c < 4; ++c) acc = wmma_bf16(alv[c], w1f[ti][c].v, acc);

            const int jc   = nn >> 5;
            const int jkk  = nn & 31;
            const int jkhi = (jkk >> 3) & 1;
            const int jidx = (jkk & 7) + 8 * ((jkk >> 4) & 1);
            #pragma unroll
            for (int i = 0; i < 8; ++i) {
                float v = fmaxf(acc[i], 0.f);
                int m = i + 8 * hi;
                apackedM[(jc * 32 + (m + 16 * jkhi)) * 16 + jidx] = f2bf(v);
            }
        }
    }
    __syncthreads();

    // logits = hid @ W2^T + b2 (N=24 padded to 32): waves 0,1 only (wave-uniform branch).
    if (w < 2) {
        v16bf amv[4];
        #pragma unroll
        for (int c = 0; c < 4; ++c) amv[c] = ldfrag(&apackedM[(c * 32 + l) * 16]);

        int   nn = w * 16 + lm;
        float bb = (nn < 24) ? b2[nn] : 0.f;
        v8f acc;
        #pragma unroll
        for (int i = 0; i < 8; ++i) acc[i] = bb;
        #pragma unroll
        for (int c = 0; c < 4; ++c) acc = wmma_bf16(amv[c], w2f[c].v, acc);

        if (nn < 24) {
            #pragma unroll
            for (int i = 0; i < 8; ++i)
                out[(size_t)(b0 + i + 8 * hi) * 24 + nn] = acc[i];
        }
    }
}

extern "C" void kernel_launch(void* const* d_in, const int* in_sizes, int n_in,
                              void* d_out, int out_size, void* d_ws, size_t ws_size,
                              hipStream_t stream) {
    (void)in_sizes; (void)n_in; (void)out_size; (void)ws_size;

    const float* x        = (const float*)d_in[0];
    const float* Wih_l0_f = (const float*)d_in[1];
    const float* Whh_l0_f = (const float*)d_in[2];
    const float* bih_l0_f = (const float*)d_in[3];
    const float* bhh_l0_f = (const float*)d_in[4];
    const float* Wih_l0_b = (const float*)d_in[5];
    const float* Whh_l0_b = (const float*)d_in[6];
    const float* bih_l0_b = (const float*)d_in[7];
    const float* bhh_l0_b = (const float*)d_in[8];
    const float* Wih_l1_f = (const float*)d_in[9];
    const float* Whh_l1_f = (const float*)d_in[10];
    const float* bih_l1_f = (const float*)d_in[11];
    const float* bhh_l1_f = (const float*)d_in[12];
    const float* Wih_l1_b = (const float*)d_in[13];
    const float* bih_l1_b = (const float*)d_in[15];
    const float* bhh_l1_b = (const float*)d_in[16];
    const float* W1       = (const float*)d_in[17];
    const float* b1       = (const float*)d_in[18];
    const float* W2       = (const float*)d_in[19];
    const float* b2       = (const float*)d_in[20];

    // Workspace: h0 bf16 [256][2048][128] (128 MB) then h1f f32 [256][64].
    unsigned short* h0 = (unsigned short*)d_ws;
    float* h1f = (float*)((char*)d_ws + (size_t)BATCH * TSEQ * D2 * 2);
    float* out = (float*)d_out;

    k_gru_l0<<<32, 128, 0, stream>>>(x,
        Wih_l0_f, Whh_l0_f, bih_l0_f, bhh_l0_f,
        Wih_l0_b, Whh_l0_b, bih_l0_b, bhh_l0_b, h0);

    k_gru_l1f<<<16, 128, 0, stream>>>(h0, Wih_l1_f, Whh_l1_f, bih_l1_f, bhh_l1_f, h1f);

    k_head<<<16, 128, 0, stream>>>(h0, h1f, Wih_l1_b, bih_l1_b, bhh_l1_b,
                                   W1, b1, W2, b2, out);
}
